// DenseDilatedKnnGraph_26637387170226
// MI455X (gfx1250) — matrix operand order repaired and checked by reading, not compiled, not run on hardware
//
#include <hip/hip_runtime.h>

typedef __attribute__((ext_vector_type(16))) _Float16     v16h;
typedef __attribute__((ext_vector_type(8)))  float        v8f;
typedef __attribute__((ext_vector_type(4)))  unsigned int v4u;

namespace {

constexpr int kB = 4;
constexpr int kC = 384;
constexpr int kN = 4096;

constexpr int BM  = 128;          // block tile M
constexpr int BN  = 128;          // block tile N
constexpr int KC  = 32;           // K chunk per LDS stage (one WMMA K)
constexpr int KITERS = kC / KC;   // 12
constexpr int STR = 40;           // padded LDS row stride in halves (80B)
constexpr int NBLK = kN / BM;     // 32
constexpr int BLKS_PER_BATCH = NBLK * NBLK;  // 1024

constexpr double kPpf = 1.2815515655446004;  // norm.ppf(0.9)

union H16 { v16h h; v4u u[2]; };

// CDNA5 async memory->LDS DMA (16 bytes), tracked by ASYNCcnt.  GV mode:
// vdst = LDS byte address (low 32 bits of generic pointer), vaddr = 64-bit
// global address, saddr = off.
__device__ __forceinline__ void async_copy_b128(const _Float16* lds_dst,
                                                const _Float16* gsrc) {
  const unsigned lds_addr = (unsigned)(size_t)lds_dst;
  asm volatile("global_load_async_to_lds_b128 %0, %1, off"
               :
               : "v"(lds_addr), "v"(gsrc)
               : "memory");
}

// Pipeline stage synchronization:
//   s_wait_dscnt 0     -- my fragment ds_loads have finished reading LDS
//   s_wait_asynccnt 0  -- my DMA chunks have landed in LDS
//   split workgroup barrier -- everyone's reads done + DMA landed
// Returns an opaque 0 token.  Adding the token to every LDS address that must
// be ordered against this barrier creates a real data dependence, so the
// scheduler cannot hoist ds_loads / DMA issues across it (observed in round 2
// when the writes were only visible as opaque inline asm).
__device__ __forceinline__ unsigned stage_sync() {
  unsigned token;
  asm volatile(
      "s_wait_dscnt 0x0\n\t"
      "s_wait_asynccnt 0x0\n\t"
      "s_barrier_signal -1\n\t"
      "s_barrier_wait -1\n\t"
      "s_mov_b32 %0, 0"
      : "=s"(token)
      :
      : "memory");
  return token;
}

// ---------------------------------------------------------------------------
// Kernel 1: L2-normalize over channel dim, emit f16 xs[b][n][c] and
// x_sq[b*N+n] (f32).  x layout: (B, C, N, 1) -> x[(b*C+c)*N+n]
// ---------------------------------------------------------------------------
__global__ __launch_bounds__(256) void normalize_kernel(
    const float* __restrict__ x, _Float16* __restrict__ xs,
    float* __restrict__ xsq) {
  const int tid = blockIdx.x * 256 + threadIdx.x;     // = b*N + n, grid exact
  const int b = tid / kN;
  const int n = tid % kN;
  const float* xb = x + (size_t)b * kC * kN + n;

  float s = 0.0f;
  for (int c = 0; c < kC; ++c) {
    float v = xb[(size_t)c * kN];                     // coalesced across lanes
    s = fmaf(v, v, s);
  }
  const float nrm = sqrtf(s);
  const float inv = 1.0f / fmaxf(nrm, 1e-12f);
  xsq[tid] = s * inv * inv;                           // == |normalized|^2

  _Float16* xr = xs + (size_t)tid * kC;
  for (int c = 0; c < kC; ++c) xr[c] = (_Float16)(xb[(size_t)c * kN] * inv);
}

// ---------------------------------------------------------------------------
// GEMM core: one 128x128 tile of inner = xs . xs^T per block, WMMA f16->f32,
// double-buffered LDS fed by async global->LDS DMA (ASYNCcnt pipelined).
// PASS 0: accumulate deterministic per-block (sum, sumsq) of sim.
// PASS 1: threshold sim against precomputed thresh[b], write int32 edges.
// ---------------------------------------------------------------------------
template <int PASS>
__global__ __launch_bounds__(256) void gemm_kernel(
    const _Float16* __restrict__ xs, const float* __restrict__ xsq,
    double* __restrict__ part1, double* __restrict__ part2,
    const float* __restrict__ thresh, int* __restrict__ out) {
  __shared__ __align__(16) _Float16 As[2][BM * STR];
  __shared__ __align__(16) _Float16 Bs[2][BN * STR];
  __shared__ float  sXi[BM];
  __shared__ float  sXj[BN];
  __shared__ double wsum[8], wsum2[8];

  const int blk  = blockIdx.x;
  const int b    = blk / BLKS_PER_BATCH;
  const int r    = blk % BLKS_PER_BATCH;
  const int ibase = (r / NBLK) * BM;
  const int jbase = (r % NBLK) * BN;

  const int t    = threadIdx.x;
  const int lane = t & 31;
  const int wave = t >> 5;        // 8 waves
  const int wm   = wave >> 1;     // 0..3  -> 32-row strip
  const int wn   = wave & 1;      // 0..1  -> 64-col strip
  const int g    = lane >> 4;     // lane group (0/1)
  const int l15  = lane & 15;

  if (t < BM)                sXi[t]      = xsq[b * kN + ibase + t];
  else if (t < BM + BN)      sXj[t - BM] = xsq[b * kN + jbase + (t - BM)];

  const _Float16* gx = xs + (size_t)b * kN * kC;

  // This thread's two (row, kc) chunks of the cooperative 128x32-half tile.
  const int row0 = t >> 2;                 // q = t
  const int kc0  = (t & 3) * 8;
  const int row1 = (t + 256) >> 2;         // q = t + 256
  const int kc1  = ((t + 256) & 3) * 8;

  v8f acc[2][4] = {};
  unsigned tok = 0;  // opaque ordering token, always 0

  // Prologue: DMA stage 0 into buffer 0.
  {
    async_copy_b128(&As[0][row0 * STR + kc0],
                    gx + (size_t)(ibase + row0) * kC + kc0);
    async_copy_b128(&Bs[0][row0 * STR + kc0],
                    gx + (size_t)(jbase + row0) * kC + kc0);
    async_copy_b128(&As[0][row1 * STR + kc1],
                    gx + (size_t)(ibase + row1) * kC + kc1);
    async_copy_b128(&Bs[0][row1 * STR + kc1],
                    gx + (size_t)(jbase + row1) * kC + kc1);
  }

  for (int kk = 0; kk < KITERS; ++kk) {
    const int buf = kk & 1;
    tok |= stage_sync();  // stage kk landed everywhere; prior reads drained

    // Kick off stage kk+1 DMA into the other buffer; overlaps the WMMAs.
    // LDS destinations carry the token so the issue cannot move above the
    // barrier (buffer may still be read by other waves until then).
    if (kk + 1 < KITERS) {
      const int k0n = (kk + 1) * KC;
      const _Float16* an = &As[buf ^ 1][tok];
      const _Float16* bn = &Bs[buf ^ 1][tok];
      async_copy_b128(an + row0 * STR + kc0,
                      gx + (size_t)(ibase + row0) * kC + k0n + kc0);
      async_copy_b128(bn + row0 * STR + kc0,
                      gx + (size_t)(jbase + row0) * kC + k0n + kc0);
      async_copy_b128(an + row1 * STR + kc1,
                      gx + (size_t)(ibase + row1) * kC + k0n + kc1);
      async_copy_b128(bn + row1 * STR + kc1,
                      gx + (size_t)(jbase + row1) * kC + k0n + kc1);
    }

    // Fragment bases carry the token: ds_loads cannot hoist above the sync.
    const _Float16* ab = &As[buf][tok];
    const _Float16* bb = &Bs[buf][tok];

    // A fragment (16x32 f16): lane holds row m=l15; lane group g selects the
    // interleaved K ranges {8g..8g+7, 16+8g..16+8g+7} (ISA 7.12.2 table).
    H16 af[2];
#pragma unroll
    for (int tm = 0; tm < 2; ++tm) {
      const int m = wm * 32 + tm * 16 + l15;
      af[tm].u[0] = *(const v4u*)&ab[m * STR + 8 * g];
      af[tm].u[1] = *(const v4u*)&ab[m * STR + 16 + 8 * g];
    }
    // B fragment (32x16 f16, column-major): lane holds column n=l15; lane
    // group g selects K {8g..8g+7, 16+8g..16+8g+7} (same as A).
    H16 bf[4];
#pragma unroll
    for (int tn = 0; tn < 4; ++tn) {
      const int nr = wn * 64 + tn * 16 + l15;
      bf[tn].u[0] = *(const v4u*)&bb[nr * STR + 8 * g];
      bf[tn].u[1] = *(const v4u*)&bb[nr * STR + 16 + 8 * g];
    }
#pragma unroll
    for (int tm = 0; tm < 2; ++tm)
#pragma unroll
      for (int tn = 0; tn < 4; ++tn)
        acc[tm][tn] = __builtin_amdgcn_wmma_f32_16x16x32_f16(
            false, af[tm].h, false, bf[tn].h, (short)0, acc[tm][tn],
            false, false);
  }

  // Epilogue.  C/D layout: lanes 0-15 -> M=vgpr, N=lane; lanes 16-31 ->
  // M=vgpr+8, N=lane-16.  sXi/sXj reads carry the token (ordered after the
  // barriers that follow their producers).
  if (PASS == 0) {
    double s1 = 0.0, s2 = 0.0;
#pragma unroll
    for (int tm = 0; tm < 2; ++tm)
#pragma unroll
      for (int tn = 0; tn < 4; ++tn)
#pragma unroll
        for (int v = 0; v < 8; ++v) {
          const int m = wm * 32 + tm * 16 + g * 8 + v;
          const int n = wn * 64 + tn * 16 + l15;
          const float sim = 2.0f * acc[tm][tn][v] - sXi[tok + m] - sXj[tok + n];
          s1 += (double)sim;
          s2 += (double)sim * (double)sim;
        }
    // Deterministic reduction: wave shuffle -> LDS -> single-thread sum.
#pragma unroll
    for (int off = 16; off > 0; off >>= 1) {
      s1 += __shfl_down(s1, off, 32);
      s2 += __shfl_down(s2, off, 32);
    }
    if (lane == 0) { wsum[wave] = s1; wsum2[wave] = s2; }
    __syncthreads();
    if (t == 0) {
      double a = 0.0, c = 0.0;
#pragma unroll
      for (int w = 0; w < 8; ++w) { a += wsum[w]; c += wsum2[w]; }
      part1[b * BLKS_PER_BATCH + r] = a;
      part2[b * BLKS_PER_BATCH + r] = c;
    }
  } else {
    const float th = thresh[b];
    const size_t bnn = (size_t)kB * kN * kN;
#pragma unroll
    for (int tm = 0; tm < 2; ++tm)
#pragma unroll
      for (int tn = 0; tn < 4; ++tn)
#pragma unroll
        for (int v = 0; v < 8; ++v) {
          const int m = wm * 32 + tm * 16 + g * 8 + v;
          const int n = wn * 64 + tn * 16 + l15;
          const int i = ibase + m;
          const int j = jbase + n;
          const float sim = 2.0f * acc[tm][tn][v] - sXi[tok + m] - sXj[tok + n];
          const bool adj = sim > th;
          const size_t off = ((size_t)(b * kN + i)) * kN + j;
          out[off]       = adj ? (b * kN + i) : -1;
          out[bnn + off] = adj ? (b * kN + j) : -1;
        }
  }
}

// ---------------------------------------------------------------------------
// Kernel 3: fixed-order reduction of block partials; thresh = mean + ppf*std.
// (sim - mean)/std > ppf  <=>  sim > mean + ppf*std   (std > 0)
// ---------------------------------------------------------------------------
__global__ void finalize_kernel(const double* __restrict__ part1,
                                const double* __restrict__ part2,
                                float* __restrict__ thresh) {
  const int b = threadIdx.x;
  if (b >= kB) return;
  double s1 = 0.0, s2 = 0.0;
  for (int q = 0; q < BLKS_PER_BATCH; ++q) {
    s1 += part1[b * BLKS_PER_BATCH + q];
    s2 += part2[b * BLKS_PER_BATCH + q];
  }
  const double M = (double)kN * (double)kN;
  const double mean = s1 / M;
  const double var = (s2 - s1 * s1 / M) / (M - 1.0);  // unbiased (ddof=1)
  const double sd = sqrt(var > 0.0 ? var : 0.0);
  thresh[b] = (float)(mean + kPpf * sd);
}

}  // namespace

extern "C" void kernel_launch(void* const* d_in, const int* in_sizes, int n_in,
                              void* d_out, int out_size, void* d_ws,
                              size_t ws_size, hipStream_t stream) {
  (void)in_sizes; (void)n_in; (void)out_size; (void)ws_size;
  const float* x = (const float*)d_in[0];
  int* out = (int*)d_out;

  // Workspace layout (all 16B-aligned): ~12.7 MB total.
  char* ws = (char*)d_ws;
  size_t o = 0;
  _Float16* xs  = (_Float16*)(ws + o); o += (size_t)kB * kN * kC * sizeof(_Float16);
  float*    xsq = (float*)(ws + o);    o += (size_t)kB * kN * sizeof(float);
  double*   p1  = (double*)(ws + o);   o += (size_t)kB * BLKS_PER_BATCH * sizeof(double);
  double*   p2  = (double*)(ws + o);   o += (size_t)kB * BLKS_PER_BATCH * sizeof(double);
  float*    th  = (float*)(ws + o);    o += (size_t)kB * sizeof(float);

  normalize_kernel<<<kB * kN / 256, 256, 0, stream>>>(x, xs, xsq);

  const dim3 grid(kB * BLKS_PER_BATCH);  // 4096 blocks
  gemm_kernel<0><<<grid, 256, 0, stream>>>(xs, xsq, p1, p2, th, out);
  finalize_kernel<<<1, 32, 0, stream>>>(p1, p2, th);
  gemm_kernel<1><<<grid, 256, 0, stream>>>(xs, xsq, p1, p2, th, out);
}
